// FullAttention_64613488001701
// MI455X (gfx1250) — matrix operand order repaired and checked by reading, not compiled, not run on hardware
//
#include <hip/hip_runtime.h>

// ---------------------------------------------------------------------------
// MI455X (gfx1250) fused attention, bf16 WMMA + Tensor Data Mover staging
// ---------------------------------------------------------------------------
typedef __bf16 bf16;
typedef __attribute__((ext_vector_type(16))) __bf16 v16bf;
typedef __attribute__((ext_vector_type(8)))  __bf16 v8bf;
typedef __attribute__((ext_vector_type(2)))  __bf16 v2bf;
typedef __attribute__((ext_vector_type(8)))  float  v8f;
typedef __attribute__((ext_vector_type(4)))  float  v4f;
typedef unsigned int u32;
typedef __attribute__((ext_vector_type(4))) u32 v4u;
typedef __attribute__((ext_vector_type(8))) int v8i;
typedef __attribute__((ext_vector_type(4))) int v4i;

#define B_    32
#define L1_   512
#define L2_   512
#define FULL_ 1024
#define HID_  1024
#define NLVL_ 4
#define DH_   256      // head dim = HIDDEN / NUM_LEVEL

#if defined(__has_builtin)
#if __has_builtin(__builtin_amdgcn_tensor_load_to_lds) && \
    __has_builtin(__builtin_amdgcn_s_wait_tensorcnt)
#define HAS_TDM 1
#endif
#endif
#ifndef HAS_TDM
#define HAS_TDM 0
#endif

// ---- Tensor Data Mover: 2-D bf16 tile load Global->LDS (ISA 08 §8) --------
#if HAS_TDM
__device__ __forceinline__ void tdm_load_2d_bf16(u32 lds_byte_addr, const void* gptr,
                                                 u32 rows, u32 rowlen_elems,
                                                 u32 row_stride_elems) {
  const unsigned long long ga = (unsigned long long)(size_t)gptr;
  v4u g0;
  g0[0] = 1u;                                   // count=1, user descriptor
  g0[1] = lds_byte_addr;                        // lds_addr
  g0[2] = (u32)ga;                              // global_addr[31:0]
  g0[3] = (u32)(ga >> 32) | 0x80000000u;        // global_addr[56:32] | type=2
  v8i g1;
  g1[0] = (int)(1u << 16);                      // wg_mask=0, data_size=1 (2B)
  g1[1] = (int)((rowlen_elems & 0xFFFFu) << 16);            // tensor_dim0 lo16
  g1[2] = (int)(((rowlen_elems >> 16) & 0xFFFFu) |
                ((rows & 0xFFFFu) << 16));                  // dim0 hi | dim1 lo
  g1[3] = (int)(((rows >> 16) & 0xFFFFu) |
                ((rowlen_elems & 0xFFFFu) << 16));          // dim1 hi | tile_dim0
  g1[4] = (int)(rows & 0xFFFFu);                            // tile_dim1 (dim2=0)
  g1[5] = (int)row_stride_elems;                            // dim0_stride lo32
  g1[6] = 0;
  g1[7] = 0;
  const v4i z4 = {0, 0, 0, 0};
#if __clang_major__ >= 23
  const v8i z8 = {0, 0, 0, 0, 0, 0, 0, 0};
  __builtin_amdgcn_tensor_load_to_lds(g0, g1, z4, z4, z8, 0);
#else
  __builtin_amdgcn_tensor_load_to_lds(g0, g1, z4, z4, 0);
#endif
}
__device__ __forceinline__ void tdm_wait_all() {
  __builtin_amdgcn_s_wait_tensorcnt(0);
}
#endif

// ---- wave32 fragment loaders (CDNA5 ISA 7.12.2 layouts) -------------------
// A 16x32 bf16: lane m = lane&15; hi = lane>>4; K = {8*hi+0..7, 16+8*hi+0..7}
__device__ __forceinline__ v16bf lds_a16x32(const bf16* base, int ld) {
  const int lane = threadIdx.x & 31;
  const int m = lane & 15, hi = lane >> 4;
  const bf16* p = base + m * ld + 8 * hi;
  v8bf lo = *(const v8bf*)(p);
  v8bf hh = *(const v8bf*)(p + 16);
  return __builtin_shufflevector(lo, hh, 0,1,2,3,4,5,6,7,8,9,10,11,12,13,14,15);
}
// B 32x16 bf16, stored transposed in LDS as Bt[n][k]: lane n = lane&15;
// K = {8*hi+0..7, 16+8*hi+0..7} (same order as A)
__device__ __forceinline__ v16bf lds_b32x16(const bf16* base, int ld) {
  const int lane = threadIdx.x & 31;
  const int n = lane & 15, hi = lane >> 4;
  const bf16* p = base + n * ld + 8 * hi;
  v8bf lo = *(const v8bf*)(p);
  v8bf hh = *(const v8bf*)(p + 16);
  return __builtin_shufflevector(lo, hh, 0,1,2,3,4,5,6,7,8,9,10,11,12,13,14,15);
}
__device__ __forceinline__ v8f wmma_bf16(v16bf a, v16bf b, v8f c) {
  return __builtin_amdgcn_wmma_f32_16x16x32_bf16(false, a, false, b,
                                                 (short)0, c, false, false);
}

// ---------------------------------------------------------------------------
// Kernel 2 (placed first so the disasm snippet shows it):
// fused flash attention per (b, level, 128-query block).
// 8 waves; wave owns 16 query rows -> softmax is wave-local (16-lane halves).
// Q/K tiles staged by Tensor Data Mover (per-wave row slices); V converted
// fp32->bf16 and transposed by VALU (TDM cannot convert).
// LDS: Q 128x256 | K 64x256 | V^T 256x72(pad) | P 8x16x64   (148 KB dynamic)
// ---------------------------------------------------------------------------
#define QBLK 128
#define KCH  64
#define VLD  72

__global__ __launch_bounds__(256)
void attn_kernel(const bf16* __restrict__ qk, const bf16* __restrict__ kk,
                 const float* __restrict__ x2, const unsigned char* __restrict__ mask,
                 float* __restrict__ out) {
  extern __shared__ char smem[];
  bf16* Qs = (bf16*)smem;                 // QBLK x DH_
  bf16* Ks = Qs + QBLK * DH_;             // KCH  x DH_
  bf16* Vt = Ks + KCH * DH_;              // DH_  x VLD   (V transposed, padded)
  bf16* Ps = Vt + DH_ * VLD;              // 8 waves x 16 x KCH

  const int tid = threadIdx.x, lane = tid & 31, wave = tid >> 5;
  const int hi = lane >> 4, nc = lane & 15;
  const int qb = blockIdx.x * QBLK;
  const int lvl = blockIdx.y;
  const int b = blockIdx.z;

  const bf16* qsrc = qk + ((size_t)b * L1_ + qb) * HID_ + lvl * DH_;
#if HAS_TDM
  const u32 QsOff = (u32)(size_t)Qs;
  const u32 KsOff = (u32)(size_t)Ks;
  // stage Q block: one TDM per wave, 16 rows of 256 bf16 (row stride 1024)
  tdm_load_2d_bf16(QsOff + (u32)(wave * 16 * DH_) * 2u,
                   qsrc + (size_t)wave * 16 * HID_, 16, DH_, HID_);
#else
  for (int i = tid; i < QBLK * DH_ / 8; i += 256) {
    const int r = i >> 5, c = (i & 31) * 8;
    *(v8bf*)(Qs + r * DH_ + c) = *(const v8bf*)(qsrc + (size_t)r * HID_ + c);
  }
#endif

  v8f  Oacc[16] = {};
  float mrow[8], lrow[8];
#pragma unroll
  for (int j = 0; j < 8; ++j) { mrow[j] = -3.0e38f; lrow[j] = 0.0f; }

  bf16* Pw = Ps + wave * 16 * KCH;
  const bf16* qbase = Qs + wave * 16 * DH_;

  for (int k0 = 0; k0 < L2_; k0 += KCH) {
    __syncthreads();  // previous-iteration LDS reads done before restaging
    const bf16* ksrc = kk + ((size_t)b * L2_ + k0) * HID_ + lvl * DH_;
    const float* vsrc = x2 + ((size_t)b * L2_ + k0) * HID_ + lvl * DH_;
#if HAS_TDM
    // stage K chunk: one TDM per wave, 8 rows of 256 bf16
    tdm_load_2d_bf16(KsOff + (u32)(wave * 8 * DH_) * 2u,
                     ksrc + (size_t)wave * 8 * HID_, 8, DH_, HID_);
#else
    for (int i = tid; i < KCH * DH_ / 8; i += 256) {
      const int r = i >> 5, c = (i & 31) * 8;
      *(v8bf*)(Ks + r * DH_ + c) = *(const v8bf*)(ksrc + (size_t)r * HID_ + c);
    }
#endif
    // stage V transposed (+convert): Vt[d][key], pack 2 keys per b32 store
    for (int i = tid; i < (KCH / 2) * DH_; i += 256) {
      const int r = (i / DH_) * 2, c = i % DH_;  // consecutive tid -> consec c
      v2bf pk;
      pk[0] = (bf16)vsrc[(size_t)r * HID_ + c];
      pk[1] = (bf16)vsrc[(size_t)(r + 1) * HID_ + c];
      *(v2bf*)(Vt + c * VLD + r) = pk;
    }
    // prefetch next chunk's K/V source lines while this chunk computes
    if (k0 + KCH < L2_) {
      __builtin_prefetch((const char*)(ksrc + (size_t)(KCH + (tid >> 2)) * HID_) +
                             (tid & 3) * 128, 0, 3);
      __builtin_prefetch((const char*)(vsrc + (size_t)(KCH + (tid >> 2)) * HID_) +
                             (tid & 3) * 256, 0, 3);
    }
#if HAS_TDM
    tdm_wait_all();      // TDM LDS writes visible to this wave
#endif
    __syncthreads();     // ... and to all waves

    // S = Q(16 x 256) . K^T  -> four 16x16 tiles per wave
    v8f Sacc[4] = {};
#pragma unroll 2
    for (int dk = 0; dk < DH_; dk += 32) {
      v16bf a = lds_a16x32(qbase + dk, DH_);
#pragma unroll
      for (int t = 0; t < 4; ++t) {
        v16bf kb = lds_b32x16(Ks + (t * 16) * DH_ + dk, DH_);
        Sacc[t] = wmma_bf16(a, kb, Sacc[t]);
      }
    }
    // mask + online softmax (row stats live in one 16-lane half per slot j)
    bool msk[4];
#pragma unroll
    for (int t = 0; t < 4; ++t)
      msk[t] = mask[(size_t)b * L2_ + k0 + t * 16 + nc] != 0;
#pragma unroll
    for (int j = 0; j < 8; ++j) {
      float cm = -3.0e38f;
#pragma unroll
      for (int t = 0; t < 4; ++t) {
        float s = msk[t] ? -3.0e38f : Sacc[t][j];
        Sacc[t][j] = s;
        cm = fmaxf(cm, s);
      }
#pragma unroll
      for (int off = 1; off < 16; off <<= 1)
        cm = fmaxf(cm, __shfl_xor(cm, off, 32));
      const float mnew = fmaxf(mrow[j], cm);
      const float scale = __expf(mrow[j] - mnew);
      mrow[j] = mnew;
      float rs = 0.0f;
#pragma unroll
      for (int t = 0; t < 4; ++t) {
        float p = __expf(Sacc[t][j] - mnew);
        Sacc[t][j] = p;
        rs += p;
      }
#pragma unroll
      for (int off = 1; off < 16; off <<= 1)
        rs += __shfl_xor(rs, off, 32);
      lrow[j] = lrow[j] * scale + rs;
#pragma unroll
      for (int td = 0; td < 16; ++td)
        Oacc[td][j] = Oacc[td][j] * scale;
    }
    // P -> LDS (wave-private), C-layout scatter
#pragma unroll
    for (int j = 0; j < 8; ++j)
#pragma unroll
      for (int t = 0; t < 4; ++t)
        Pw[(j + 8 * hi) * KCH + t * 16 + nc] = (bf16)Sacc[t][j];
    // O += P(16x64) . V(64x256), contraction split 2x32
#pragma unroll
    for (int ks = 0; ks < KCH; ks += 32) {
      v16bf a = lds_a16x32(Pw + ks, KCH);
#pragma unroll
      for (int td = 0; td < 16; ++td) {
        v16bf vb = lds_b32x16(Vt + (td * 16) * VLD + ks, VLD);
        Oacc[td] = wmma_bf16(a, vb, Oacc[td]);
      }
    }
  }
  // epilogue: normalize, write fp32 output
#pragma unroll
  for (int j = 0; j < 8; ++j) {
    const float inv = 1.0f / lrow[j];  // row 0 always unmasked -> lrow > 0
    const int q = qb + wave * 16 + j + 8 * hi;
    float* orow = out + ((size_t)b * L1_ + q) * HID_ + lvl * DH_;
#pragma unroll
    for (int td = 0; td < 16; ++td)
      orow[td * 16 + nc] = Oacc[td][j] * inv;
  }
}

// ---------------------------------------------------------------------------
// Kernel 1: keys = relu(x_att @ W^T) [* final_v for x2], output bf16.
// Block = 128x128 tile, 8 waves (4 M x 2 N), wave = 32x64 (8 accumulators).
// K-step 64 with double-buffered LDS: staging of step k+1 overlaps the 16
// WMMAs of step k. (fp32->bf16 convert at staging, so no TDM here.)
// ---------------------------------------------------------------------------
#define KB_K 64

__global__ __launch_bounds__(256)
void keys_kernel(const float* __restrict__ x1_att, const float* __restrict__ x2_att,
                 const float* __restrict__ W, const float* __restrict__ final_v,
                 bf16* __restrict__ k1out, bf16* __restrict__ k2out) {
  __shared__ bf16 Als[2][128 * KB_K];
  __shared__ bf16 Bls[2][128 * KB_K];
  const int tid = threadIdx.x, lane = tid & 31, wave = tid >> 5;
  const int which = blockIdx.z;
  const float* A = which ? x2_att : x1_att;
  bf16* Out = which ? k2out : k1out;
  const int m0 = blockIdx.x * 128, n0 = blockIdx.y * 128;
  const int wm = wave >> 1, wn = wave & 1;            // 4 x 2 wave grid
  v8f acc[2][4] = {};
  const int sr = tid >> 1, sh = tid & 1;              // staging: row, 32-el half

  auto stage = [&](int buf, int k0) {
    const float* ap = A + (size_t)(m0 + sr) * FULL_ + k0 + sh * 32;
    const float* bp = W + (size_t)(n0 + sr) * FULL_ + k0 + sh * 32;
#pragma unroll
    for (int q = 0; q < 4; ++q) {                     // 4 x v8bf per matrix
      v4f a0 = *(const v4f*)(ap + q * 8);
      v4f a1 = *(const v4f*)(ap + q * 8 + 4);
      v4f b0 = *(const v4f*)(bp + q * 8);
      v4f b1 = *(const v4f*)(bp + q * 8 + 4);
      v8bf av, bv;
#pragma unroll
      for (int e = 0; e < 4; ++e) {
        av[e] = (bf16)a0[e]; av[e + 4] = (bf16)a1[e];
        bv[e] = (bf16)b0[e]; bv[e + 4] = (bf16)b1[e];
      }
      *(v8bf*)(&Als[buf][sr * KB_K + sh * 32 + q * 8]) = av;
      *(v8bf*)(&Bls[buf][sr * KB_K + sh * 32 + q * 8]) = bv;
    }
  };

  stage(0, 0);
  int cur = 0;
  for (int kk = 0; kk < FULL_ / KB_K; ++kk) {
    __syncthreads();                    // staging of buf[cur] complete
    if (kk + 1 < FULL_ / KB_K)
      stage(cur ^ 1, (kk + 1) * KB_K);  // overlap next staging with WMMAs
    const bf16* Ab = Als[cur];
    const bf16* Bb = Bls[cur];
#pragma unroll
    for (int dk = 0; dk < KB_K; dk += 32)
#pragma unroll
      for (int i = 0; i < 2; ++i) {
        v16bf a = lds_a16x32(Ab + (wm * 32 + i * 16) * KB_K + dk, KB_K);
#pragma unroll
        for (int t = 0; t < 4; ++t) {
          v16bf b = lds_b32x16(Bb + (wn * 64 + t * 16) * KB_K + dk, KB_K);
          acc[i][t] = wmma_bf16(a, b, acc[i][t]);
        }
      }
    cur ^= 1;
  }
  // epilogue: relu (+ final_v for x2), store bf16
  const int hi = lane >> 4, nc = lane & 15;
#pragma unroll
  for (int i = 0; i < 2; ++i)
#pragma unroll
    for (int t = 0; t < 4; ++t) {
      const int ncol = n0 + wn * 64 + t * 16 + nc;
      const float fv = which ? final_v[ncol] : 1.0f;
#pragma unroll
      for (int j = 0; j < 8; ++j) {
        const int mrow = m0 + wm * 32 + i * 16 + j + 8 * hi;
        float v = acc[i][t][j];
        v = v > 0.0f ? v : 0.0f;
        Out[(size_t)mrow * HID_ + ncol] = (bf16)(v * fv);
      }
    }
}

// ---------------------------------------------------------------------------
extern "C" void kernel_launch(void* const* d_in, const int* in_sizes, int n_in,
                              void* d_out, int out_size, void* d_ws, size_t ws_size,
                              hipStream_t stream) {
  const float* x1_att = (const float*)d_in[0];
  const float* x2_att = (const float*)d_in[1];
  const float* x2     = (const float*)d_in[2];
  const unsigned char* x2_mask = (const unsigned char*)d_in[3];
  const float* W       = (const float*)d_in[4];
  const float* final_v = (const float*)d_in[5];
  float* out = (float*)d_out;

  bf16* k1 = (bf16*)d_ws;                                  // 32 MB
  bf16* k2 = k1 + (size_t)B_ * L1_ * HID_;                 // 32 MB

  dim3 gk(16384 / 128, HID_ / 128, 2);
  keys_kernel<<<gk, 256, 0, stream>>>(x1_att, x2_att, W, final_v, k1, k2);

  const size_t smem = (size_t)(QBLK * DH_ + KCH * DH_ + DH_ * VLD + 8 * 16 * KCH)
                      * sizeof(bf16);                      // 151552 B
  (void)hipFuncSetAttribute((const void*)attn_kernel,
                            hipFuncAttributeMaxDynamicSharedMemorySize, (int)smem);
  dim3 ga(L1_ / QBLK, NLVL_, B_);
  attn_kernel<<<ga, 256, smem, stream>>>(k1, k2, x2, x2_mask, out);
}
